// CustomUltrasoundViT_49065706389836
// MI455X (gfx1250) — matrix operand order repaired and checked by reading, not compiled, not run on hardware
//
#include <hip/hip_runtime.h>
#include <hip/hip_bf16.h>
#include <cstdint>
#include <cstddef>

// ---------------- model constants ----------------
#define DMODEL 768
#define NHEAD  12
#define DHEAD  64
#define DMLP   3072
#define NTOKS  1025
#define BATCH  4
#define MROWS  4100            // BATCH * NTOKS
#define SPAD   1056            // kv length padded to multiple of 32 (33 * 32)
#define NBH    48              // BATCH * NHEAD
#define LDH    40              // LDS half-word row stride (80B: 16B aligned, conflict-free)

typedef __attribute__((ext_vector_type(16))) _Float16 v16h;
typedef __attribute__((ext_vector_type(8)))  _Float16 h8v;
typedef __attribute__((ext_vector_type(8)))  float    v8f;
typedef __attribute__((ext_vector_type(4))) unsigned int u32x4;
typedef __attribute__((ext_vector_type(4))) int          i32x4;
typedef __attribute__((ext_vector_type(8))) int          i32x8;

enum { F_GELU = 1, F_SQK = 2, F_SVT = 4 };

// Tensor Data Mover availability (ROCm7.2 clang-22: 5-arg; clang-23: 6-arg)
#if defined(__has_builtin)
#if __has_builtin(__builtin_amdgcn_tensor_load_to_lds) && \
    __has_builtin(__builtin_amdgcn_s_wait_tensorcnt)
#define USE_TDM 1
#endif
#endif
#ifndef USE_TDM
#define USE_TDM 0
#endif

__device__ __forceinline__ v8f wmma16(v16h a, v16h b, v8f c) {
  // D = A(16x32 f16) * B(32x16 f16) + C(16x16 f32)
  return __builtin_amdgcn_wmma_f32_16x16x32_f16(false, a, false, b, (short)0, c,
                                                false, false);
}
__device__ __forceinline__ v16h mk16(h8v lo, h8v hi) {
  v16h r;
#pragma unroll
  for (int i = 0; i < 8; ++i) { r[i] = lo[i]; r[i + 8] = hi[i]; }
  return r;
}
__device__ __forceinline__ float gelu_exact(float x) {
  return 0.5f * x * (1.f + erff(x * 0.70710678118654752f));
}

#if USE_TDM
// Issue one TDM 2-D tile load: 32 f16 cols x 128 rows from a [rows x strideElems]
// f16 tensor into LDS at ldsAddr. TDM hardware padding inserts 16B after every
// 64B row so the LDS image has the bank-conflict-free LDH(=40-half) row stride.
// Rows past rowsRemaining are zero-filled by TDM OOB handling (M-tail clamp).
__device__ __forceinline__ void tdm_load_tile2d(const void* gsrc, unsigned ldsAddr,
                                                unsigned rowsRemaining,
                                                unsigned strideElems)
{
  unsigned long long ga = (unsigned long long)(uintptr_t)gsrc;
  u32x4 g0;
  g0[0] = 1u;                                          // count=1 (valid user D#)
  g0[1] = ldsAddr;                                     // lds_addr[31:0]
  g0[2] = (unsigned)ga;                                // global_addr[31:0]
  g0[3] = (unsigned)((ga >> 32) & 0x01ffffffu) | (2u << 30);  // addr[56:32]|type=2
  const unsigned td0 = 32u;                            // tensor_dim0 == tile_dim0
  i32x8 g1;
  g1[0] = (int)((1u << 16)        // data_size = 2 bytes
              | (1u << 20)        // pad_enable
              | (3u << 22)        // pad_interval: 16 DWORDs (64B row)
              | (3u << 25));      // pad_amount:   4 DWORDs (16B pad)
  g1[1] = (int)((td0 & 0xffffu) << 16);                // abar=0 | tensor_dim0 lo
  g1[2] = (int)(((td0 >> 16) & 0xffffu) | ((rowsRemaining & 0xffffu) << 16));
  g1[3] = (int)(((rowsRemaining >> 16) & 0xffffu) | (32u << 16)); // tile_dim0=32
  g1[4] = (int)128;                                    // tile_dim1=128, tile_dim2=0
  g1[5] = (int)strideElems;                            // tensor_dim0_stride lo
  g1[6] = 0;                                           // stride0 hi | stride1 lo
  g1[7] = 0;
  i32x4 gz = {0, 0, 0, 0};
#if __clang_major__ >= 23
  i32x8 gz8 = {0, 0, 0, 0, 0, 0, 0, 0};
  __builtin_amdgcn_tensor_load_to_lds(g0, g1, gz, gz, gz8, 0);
#else
  __builtin_amdgcn_tensor_load_to_lds(g0, g1, gz, gz, 0);
#endif
}
#endif

// ---------------- generic WMMA GEMM ----------------
// C[M,N] = A[M,K](f16) * Bt[N,K](f16)^T, epilogue: +bias, gelu, +resid,
// store f32 row-major and/or f16 (row-major or QKV scatter).
// TDM path: wave0 drives double-buffered tensor_load_to_lds DMA, all waves
// compute; fallback path: cooperative b128 staging.
__global__ __launch_bounds__(256) void k_gemm(
    const _Float16* __restrict__ A, const _Float16* __restrict__ Bt,
    const float* __restrict__ bias, const float* resid,
    float* o32, _Float16* o16, int M, int N, int K, int flags)
{
  __shared__ _Float16 As[2][128][LDH];
  __shared__ _Float16 Bs[2][128][LDH];
  const int tid  = threadIdx.x;
  const int wave = tid >> 5, lane = tid & 31;
  const int lidx = lane & 15, hi = lane >> 4;
  const int wm = wave >> 1, wn = wave & 1;         // 4x2 wave grid -> 32x64 per wave
  const int m0 = blockIdx.y * 128, n0 = blockIdx.x * 128;

  v8f zero = {};
  v8f acc[2][4];
#pragma unroll
  for (int i = 0; i < 2; ++i)
#pragma unroll
    for (int j = 0; j < 4; ++j) acc[i][j] = zero;

  int buf = 0;
#if USE_TDM
  const bool issuer = (tid < 32);                  // one TDM producer wave
  const unsigned ldsA0 = (unsigned)(uintptr_t)&As[0][0][0];
  const unsigned ldsA1 = (unsigned)(uintptr_t)&As[1][0][0];
  const unsigned ldsB0 = (unsigned)(uintptr_t)&Bs[0][0][0];
  const unsigned ldsB1 = (unsigned)(uintptr_t)&Bs[1][0][0];
  if (issuer) {
    tdm_load_tile2d(A  + (size_t)m0 * K, ldsA0, (unsigned)(M - m0), (unsigned)K);
    tdm_load_tile2d(Bt + (size_t)n0 * K, ldsB0, (unsigned)(N - n0), (unsigned)K);
  }
#endif

  for (int k0 = 0; k0 < K; k0 += 32) {
#if USE_TDM
    if (issuer) __builtin_amdgcn_s_wait_tensorcnt((short)0);
    __syncthreads();                               // LDS[buf] visible to all waves
    if (issuer && (k0 + 32 < K)) {                 // overlap next tile DMA w/ math
      tdm_load_tile2d(A  + (size_t)m0 * K + (k0 + 32), buf ? ldsA0 : ldsA1,
                      (unsigned)(M - m0), (unsigned)K);
      tdm_load_tile2d(Bt + (size_t)n0 * K + (k0 + 32), buf ? ldsB0 : ldsB1,
                      (unsigned)(N - n0), (unsigned)K);
    }
#else
    // cooperative staging: 128x32 f16 tiles, 8 halves per b128 transaction
#pragma unroll
    for (int c = 0; c < 2; ++c) {
      int idx = tid + c * 256;                     // 512 chunks of 8 halves
      int row = idx >> 2, kc = (idx & 3) * 8;
      int gm = m0 + row; if (gm > M - 1) gm = M - 1;
      *(h8v*)&As[0][row][kc] = *(const h8v*)(A + (size_t)gm * K + k0 + kc);
      int gn = n0 + row; if (gn > N - 1) gn = N - 1;
      *(h8v*)&Bs[0][row][kc] = *(const h8v*)(Bt + (size_t)gn * K + k0 + kc);
    }
    __syncthreads();
#endif
    if (k0 + 64 < K) {                             // global_prefetch_b8 ahead
      int pr = m0 + (tid >> 1); if (pr > M - 1) pr = M - 1;
      __builtin_prefetch(A + (size_t)pr * K + k0 + 64, 0, 1);
      int pn = n0 + (tid >> 1); if (pn > N - 1) pn = N - 1;
      __builtin_prefetch(Bt + (size_t)pn * K + k0 + 64, 0, 1);
    }
    // A fragments: lane<16 -> K 0..7 / 16..23 ; lane>=16 -> K 8..15 / 24..31
    v16h af[2], bf[4];
    const int kb = hi ? 8 : 0;
#pragma unroll
    for (int mi = 0; mi < 2; ++mi) {
      int r = wm * 32 + mi * 16 + lidx;
      af[mi] = mk16(*(const h8v*)&As[buf][r][kb], *(const h8v*)&As[buf][r][kb + 16]);
    }
    // B fragments: lane = N col, lane<16 -> K 0..7 / 16..23 ; lane>=16 -> K 8..15 / 24..31
    const int kb2 = hi ? 8 : 0;
#pragma unroll
    for (int nj = 0; nj < 4; ++nj) {
      int r = wn * 64 + nj * 16 + lidx;
      bf[nj] = mk16(*(const h8v*)&Bs[buf][r][kb2], *(const h8v*)&Bs[buf][r][kb2 + 16]);
    }
#pragma unroll
    for (int mi = 0; mi < 2; ++mi)
#pragma unroll
      for (int nj = 0; nj < 4; ++nj)
        acc[mi][nj] = wmma16(af[mi], bf[nj], acc[mi][nj]);
    __syncthreads();                               // done reading LDS[buf]
#if USE_TDM
    buf ^= 1;
#endif
  }

  // epilogue (C layout: vgpr r, lane<16 -> M=r ; lane>=16 -> M=r+8 ; N=lane&15)
  const int mwb = m0 + wm * 32;
  const int nwb = n0 + wn * 64;
#pragma unroll
  for (int mi = 0; mi < 2; ++mi) {
#pragma unroll
    for (int nj = 0; nj < 4; ++nj) {
      v8f c = acc[mi][nj];
      int n = nwb + nj * 16 + lidx;
      float bvv = bias ? bias[n] : 0.f;
#pragma unroll
      for (int r = 0; r < 8; ++r) {
        int m = mwb + mi * 16 + hi * 8 + r;
        if (m < M) {
          float v = c[r] + bvv;
          if (flags & F_GELU) v = gelu_exact(v);
          if (resid) v += resid[(size_t)m * N + n];
          if (o32) o32[(size_t)m * N + n] = v;
          if (o16) {
            size_t oi;
            if (flags & (F_SQK | F_SVT)) {
              int b = m / NTOKS, s = m - b * NTOKS;
              int hh = n >> 6, hd = n & 63;
              if (flags & F_SQK) oi = ((size_t)(b * NHEAD + hh) * SPAD + s) * DHEAD + hd;
              else               oi = ((size_t)(b * NHEAD + hh) * DHEAD + hd) * SPAD + s;
            } else {
              oi = (size_t)m * N + n;
            }
            o16[oi] = (_Float16)v;
          }
        }
      }
    }
  }
}

// ---------------- flash attention (wave32, barrier-free) ----------------
// Q,K: [BH][SPAD][64] f16 ; Vt: [BH][64][SPAD] f16 (pads zeroed) ;
// ctx: [B*NTOKS][768] f16. One wave owns 16 query rows; all softmax state in
// per-wave LDS (same-wave DS ops are in-order -> no barriers needed).
__global__ __launch_bounds__(256) void k_attention(
    const _Float16* __restrict__ Q, const _Float16* __restrict__ Kx,
    const _Float16* __restrict__ Vt, _Float16* __restrict__ ctx, int S)
{
  __shared__ float    Sbuf[8][16][33];
  __shared__ _Float16 Pbuf[8][16][LDH];
  __shared__ float    mrun[8][16], srun[8][16], cfb[8][16];

  const int wave = threadIdx.x >> 5, lane = threadIdx.x & 31;
  const int lidx = lane & 15, hi = lane >> 4;
  const int bh = blockIdx.y;
  const int b = bh / NHEAD, h = bh - b * NHEAD;
  const int q0 = blockIdx.x * 128 + wave * 16;
  if (q0 >= S) return;                             // no block barriers below

  const _Float16* Qb = Q  + (size_t)bh * SPAD * DHEAD;
  const _Float16* Kb = Kx + (size_t)bh * SPAD * DHEAD;
  const _Float16* Vb = Vt + (size_t)bh * DHEAD * SPAD;

  int qr = q0 + lidx; if (qr > S - 1) qr = S - 1;
  const int kb  = hi ? 8 : 0;
  const int kb2 = hi ? 8 : 0;

  v16h aq[2];                                      // Q pre-scaled by 1/sqrt(HD)
#pragma unroll
  for (int c = 0; c < 2; ++c) {
    const _Float16* p = Qb + (size_t)qr * DHEAD + c * 32 + kb;
    v16h t = mk16(*(const h8v*)p, *(const h8v*)(p + 16));
#pragma unroll
    for (int i = 0; i < 16; ++i) t[i] = t[i] * (_Float16)0.125f;
    aq[c] = t;
  }

  if (lane < 16) { mrun[wave][lane] = -1e30f; srun[wave][lane] = 0.f; }
  v8f zero = {};
  v8f o[4];
#pragma unroll
  for (int j = 0; j < 4; ++j) o[j] = zero;

  for (int kv0 = 0; kv0 < SPAD; kv0 += 32) {
    // ---- scores S = (Q*scale) @ K^T  (two 16-col kv subtiles) ----
    v8f sc[2];
#pragma unroll
    for (int n = 0; n < 2; ++n) {
      int kvr = kv0 + n * 16 + lidx;               // lane = kv column
      v8f c = zero;
#pragma unroll
      for (int cc = 0; cc < 2; ++cc) {             // hd chunks of 32
        const _Float16* p = Kb + (size_t)kvr * DHEAD + cc * 32 + kb2;
        v16h bk = mk16(*(const h8v*)p, *(const h8v*)(p + 16));
        c = wmma16(aq[cc], bk, c);
      }
      sc[n] = c;
    }
    // spill to per-wave LDS with kv mask
#pragma unroll
    for (int n = 0; n < 2; ++n) {
      int col = n * 16 + lidx;
      bool valid = (kv0 + col) < S;
#pragma unroll
      for (int r = 0; r < 8; ++r)
        Sbuf[wave][hi * 8 + r][col] = valid ? sc[n][r] : -1e30f;
    }
    // ---- streaming softmax: lane pair (l, l+16) owns row l ----
    const int row = lidx;
    float vals[16]; float tmax = -1e30f;
#pragma unroll
    for (int i = 0; i < 16; ++i) {
      vals[i] = Sbuf[wave][row][hi * 16 + i];
      tmax = fmaxf(tmax, vals[i]);
    }
    tmax = fmaxf(tmax, __shfl_xor(tmax, 16, 32));
    float mold = mrun[wave][row];
    float mnew = fmaxf(mold, tmax);
    float cf = __expf(mold - mnew);
    float ps = 0.f;
#pragma unroll
    for (int i = 0; i < 16; ++i) {
      float e = __expf(vals[i] - mnew);
      ps += e;
      Pbuf[wave][row][hi * 16 + i] = (_Float16)e;
    }
    ps += __shfl_xor(ps, 16, 32);
    if (!hi) {
      mrun[wave][row] = mnew;
      srun[wave][row] = srun[wave][row] * cf + ps;
      cfb[wave][row]  = cf;
    }
    // rescale O accumulators
    float cf8[8];
#pragma unroll
    for (int r = 0; r < 8; ++r) cf8[r] = cfb[wave][hi * 8 + r];
#pragma unroll
    for (int j = 0; j < 4; ++j)
#pragma unroll
      for (int r = 0; r < 8; ++r) o[j][r] *= cf8[r];
    // ---- O += P @ V  (P A-frag from LDS, V B-frags contiguous from Vt) ----
    const _Float16* pp = &Pbuf[wave][lidx][kb];
    v16h pa = mk16(*(const h8v*)pp, *(const h8v*)(pp + 16));
#pragma unroll
    for (int j = 0; j < 4; ++j) {
      int hd = j * 16 + lidx;                      // lane = hd column
      const _Float16* p = Vb + (size_t)hd * SPAD + kv0 + kb2;
      v16h bvv = mk16(*(const h8v*)p, *(const h8v*)(p + 16));
      o[j] = wmma16(pa, bvv, o[j]);
    }
  }
  // normalize + store ctx[b, q, h*64+hd]
  float inv8[8];
#pragma unroll
  for (int r = 0; r < 8; ++r) inv8[r] = 1.f / srun[wave][hi * 8 + r];
#pragma unroll
  for (int j = 0; j < 4; ++j) {
    int hd = j * 16 + lidx;
#pragma unroll
    for (int r = 0; r < 8; ++r) {
      int q = q0 + hi * 8 + r;
      if (q < S)
        ctx[((size_t)b * NTOKS + q) * DMODEL + h * DHEAD + hd] =
            (_Float16)(o[j][r] * inv8[r]);
    }
  }
}

// ---------------- layernorm: f32 row -> f16 row ----------------
__global__ __launch_bounds__(256) void k_layernorm(
    const float* __restrict__ x, const float* __restrict__ g,
    const float* __restrict__ bb, _Float16* __restrict__ y)
{
  __shared__ float red[256];
  const int row = blockIdx.x, t = threadIdx.x;
  const float* xr = x + (size_t)row * DMODEL;
  float v0 = xr[t], v1 = xr[t + 256], v2 = xr[t + 512];
  red[t] = v0 + v1 + v2;
  __syncthreads();
  for (int o = 128; o > 0; o >>= 1) { if (t < o) red[t] += red[t + o]; __syncthreads(); }
  float mean = red[0] * (1.f / 768.f);
  __syncthreads();
  float d0 = v0 - mean, d1 = v1 - mean, d2 = v2 - mean;
  red[t] = d0 * d0 + d1 * d1 + d2 * d2;
  __syncthreads();
  for (int o = 128; o > 0; o >>= 1) { if (t < o) red[t] += red[t + o]; __syncthreads(); }
  float rstd = rsqrtf(red[0] * (1.f / 768.f) + 1e-5f);
  _Float16* yr = y + (size_t)row * DMODEL;
  yr[t]       = (_Float16)(d0 * rstd * g[t]       + bb[t]);
  yr[t + 256] = (_Float16)(d1 * rstd * g[t + 256] + bb[t + 256]);
  yr[t + 512] = (_Float16)(d2 * rstd * g[t + 512] + bb[t + 512]);
}

// ---------------- small utility kernels ----------------
__global__ void k_cvtT(const float* __restrict__ src, _Float16* __restrict__ dst,
                       int R, int C)   // dst[C][R] = f16(src[R][C]^T)
{
  __shared__ float tile[32][33];
  int bx = blockIdx.x * 32, by = blockIdx.y * 32;
  int tx = threadIdx.x & 31, ty = threadIdx.x >> 5;   // 8 rows per pass
  for (int i = ty; i < 32; i += 8) {
    int r = by + i, c = bx + tx;
    tile[i][tx] = (r < R && c < C) ? src[(size_t)r * C + c] : 0.f;
  }
  __syncthreads();
  for (int i = ty; i < 32; i += 8) {
    int r = bx + i, c = by + tx;
    if (r < C && c < R) dst[(size_t)r * R + c] = (_Float16)tile[tx][i];
  }
}
__global__ void k_cast16(const float* __restrict__ s, _Float16* __restrict__ d, size_t n) {
  size_t i = (size_t)blockIdx.x * 256 + threadIdx.x;
  if (i < n) d[i] = (_Float16)s[i];
}
__global__ void k_zero16(_Float16* p, size_t n) {
  size_t i = (size_t)blockIdx.x * 256 + threadIdx.x;
  if (i < n) p[i] = (_Float16)0.f;
}
__global__ void k_im2col(const float* __restrict__ x, _Float16* __restrict__ xp) {
  int idx = blockIdx.x * 256 + threadIdx.x;
  if (idx >= 4 * 1024 * 768) return;
  int k = idx % 768, rowp = idx / 768;
  int b = rowp >> 10, p = rowp & 1023;
  int i = p >> 5, j = p & 31;
  int c = k >> 8, rem = k & 255, ph = rem >> 4, pw = rem & 15;
  xp[idx] = (_Float16)x[((size_t)(b * 3 + c) * 512 + i * 16 + ph) * 512 + j * 16 + pw];
}
__global__ void k_addpos(float* h, const float* __restrict__ cls,
                         const float* __restrict__ pos) {
  int idx = blockIdx.x * 256 + threadIdx.x;
  if (idx >= MROWS * DMODEL) return;
  int d = idx % DMODEL;
  int t = (idx / DMODEL) % NTOKS;
  if (t == 0) h[idx] = cls[d] + pos[d];
  else        h[idx] += pos[(size_t)t * DMODEL + d];
}
__global__ void k_head2(const _Float16* __restrict__ hid, const float* __restrict__ w2,
                        const float* __restrict__ b2, float* out, int no) {
  int idx = blockIdx.x * 256 + threadIdx.x;
  if (idx >= MROWS * no) return;
  int row = idx / no, n = idx - row * no;
  const _Float16* hr = hid + (size_t)row * 384;
  float s = b2[n];
  for (int k = 0; k < 384; ++k) s += (float)hr[k] * w2[k * no + n];
  out[idx] = s;
}

// ---------------- host-side helpers ----------------
static void gemm(hipStream_t st, const _Float16* A, const _Float16* Bt,
                 const float* bias, const float* resid, float* o32,
                 _Float16* o16, int M, int N, int K, int flags) {
  dim3 g(N / 128, (M + 127) / 128);
  k_gemm<<<g, 256, 0, st>>>(A, Bt, bias, resid, o32, o16, M, N, K, flags);
}
static void cvtT(hipStream_t st, const float* src, _Float16* dst, int R, int C) {
  dim3 g((C + 31) / 32, (R + 31) / 32);
  k_cvtT<<<g, 256, 0, st>>>(src, dst, R, C);
}

extern "C" void kernel_launch(void* const* d_in, const int* in_sizes, int n_in,
                              void* d_out, int out_size, void* d_ws, size_t ws_size,
                              hipStream_t stream)
{
  const float* x       = (const float*)d_in[0];
  const float* conv_w  = (const float*)d_in[1];
  const float* conv_b  = (const float*)d_in[2];
  const float* cls_tok = (const float*)d_in[3];
  const float* pos_emb = (const float*)d_in[4];
  const float* ln1g = (const float*)d_in[5];
  const float* ln1b = (const float*)d_in[6];
  const float* wq = (const float*)d_in[7];   const float* bq = (const float*)d_in[8];
  const float* wk = (const float*)d_in[9];   const float* bk = (const float*)d_in[10];
  const float* wv = (const float*)d_in[11];  const float* bv = (const float*)d_in[12];
  const float* wo = (const float*)d_in[13];  const float* bo = (const float*)d_in[14];
  const float* ln2g = (const float*)d_in[15];
  const float* ln2b = (const float*)d_in[16];
  const float* w1 = (const float*)d_in[17];  const float* b1 = (const float*)d_in[18];
  const float* w2 = (const float*)d_in[19];  const float* b2 = (const float*)d_in[20];
  const float* cwq = (const float*)d_in[21]; const float* cbq = (const float*)d_in[22];
  const float* cwk = (const float*)d_in[23]; const float* cbk = (const float*)d_in[24];
  const float* cwv = (const float*)d_in[25]; const float* cbv = (const float*)d_in[26];
  const float* cwo = (const float*)d_in[27]; const float* cbo = (const float*)d_in[28];
  const float* fh_w1 = (const float*)d_in[29]; const float* fh_b1 = (const float*)d_in[30];
  const float* fh_w2 = (const float*)d_in[31]; const float* fh_b2 = (const float*)d_in[32];
  const float* vh_w1 = (const float*)d_in[33]; const float* vh_b1 = (const float*)d_in[34];
  const float* vh_w2 = (const float*)d_in[35]; const float* vh_b2 = (const float*)d_in[36];
  const float* ch_w1 = (const float*)d_in[37]; const float* ch_b1 = (const float*)d_in[38];
  const float* ch_w2 = (const float*)d_in[39]; const float* ch_b2 = (const float*)d_in[40];

  float* fascia = (float*)d_out;
  float* vein   = fascia + (size_t)MROWS * 2;
  float* clsout = vein   + (size_t)MROWS * 4;
  float* feats  = clsout + (size_t)MROWS * 3;     // [4,1025,768] f32 output

  // workspace bump allocator (~97 MB total)
  char* cur = (char*)d_ws;
  auto alloc = [&](size_t bytes) { void* p = cur; cur += (bytes + 255) & ~(size_t)255; return p; };
  float*    h     = (float*)   alloc((size_t)MROWS * DMODEL * 4);
  _Float16* hn16  = (_Float16*)alloc((size_t)MROWS * DMODEL * 2);
  _Float16* ctx16 = (_Float16*)alloc((size_t)MROWS * DMODEL * 2);
  _Float16* vf16  = (_Float16*)alloc((size_t)MROWS * DMODEL * 2);
  _Float16* ft16  = (_Float16*)alloc((size_t)MROWS * DMODEL * 2);
  _Float16* mh16  = (_Float16*)alloc((size_t)MROWS * DMLP   * 2);
  _Float16* Qb    = (_Float16*)alloc((size_t)NBH * SPAD * DHEAD * 2);
  _Float16* Kb    = (_Float16*)alloc((size_t)NBH * SPAD * DHEAD * 2);
  _Float16* Vtb   = (_Float16*)alloc((size_t)NBH * DHEAD * SPAD * 2);
  _Float16* wT    = (_Float16*)alloc((size_t)DMLP * DMODEL * 2);
  _Float16* xp16  = (_Float16*)alloc((size_t)4096 * DMODEL * 2);
  _Float16* hid16 = (_Float16*)alloc((size_t)MROWS * 384 * 2);

  // V^T buffer must be zero in the kv-pad region (P==0 there, but keep it NaN-free)
  { size_t n = (size_t)NBH * DHEAD * SPAD;
    k_zero16<<<(int)((n + 255) / 256), 256, 0, stream>>>(Vtb, n); }

  // ---- patch embedding: im2col + WMMA GEMM, output straight into h rows 1..1024 ----
  { size_t n = (size_t)4 * 1024 * 768;
    k_im2col<<<(int)((n + 255) / 256), 256, 0, stream>>>(x, xp16); }
  { size_t n = (size_t)DMODEL * DMODEL;   // conv_w is already [N=768 out][K=768]
    k_cast16<<<(int)((n + 255) / 256), 256, 0, stream>>>(conv_w, wT, n); }
  for (int b4 = 0; b4 < 4; ++b4)
    gemm(stream, xp16 + (size_t)b4 * 1024 * DMODEL, wT, conv_b, nullptr,
         h + ((size_t)b4 * NTOKS + 1) * DMODEL, nullptr, 1024, DMODEL, DMODEL, 0);
  { size_t n = (size_t)MROWS * DMODEL;
    k_addpos<<<(int)((n + 255) / 256), 256, 0, stream>>>(h, cls_tok, pos_emb); }

  const dim3 attnGrid((NTOKS + 127) / 128, NBH);

  // ---- 12 transformer blocks ----
  for (int l = 0; l < 12; ++l) {
    size_t o2 = (size_t)l * DMODEL * DMODEL;
    size_t o1 = (size_t)l * DMODEL;
    size_t om = (size_t)l * DMODEL * DMLP;
    size_t ob = (size_t)l * DMLP;
    k_layernorm<<<MROWS, 256, 0, stream>>>(h, ln1g + o1, ln1b + o1, hn16);
    cvtT(stream, wq + o2, wT, DMODEL, DMODEL);
    gemm(stream, hn16, wT, bq + o1, nullptr, nullptr, Qb, MROWS, DMODEL, DMODEL, F_SQK);
    cvtT(stream, wk + o2, wT, DMODEL, DMODEL);
    gemm(stream, hn16, wT, bk + o1, nullptr, nullptr, Kb, MROWS, DMODEL, DMODEL, F_SQK);
    cvtT(stream, wv + o2, wT, DMODEL, DMODEL);
    gemm(stream, hn16, wT, bv + o1, nullptr, nullptr, Vtb, MROWS, DMODEL, DMODEL, F_SVT);
    k_attention<<<attnGrid, 256, 0, stream>>>(Qb, Kb, Vtb, ctx16, NTOKS);
    cvtT(stream, wo + o2, wT, DMODEL, DMODEL);
    gemm(stream, ctx16, wT, bo + o1, h, h, nullptr, MROWS, DMODEL, DMODEL, 0);
    k_layernorm<<<MROWS, 256, 0, stream>>>(h, ln2g + o1, ln2b + o1, hn16);
    cvtT(stream, w1 + om, wT, DMODEL, DMLP);
    gemm(stream, hn16, wT, b1 + ob, nullptr, nullptr, mh16, MROWS, DMLP, DMODEL, F_GELU);
    cvtT(stream, w2 + om, wT, DMLP, DMODEL);
    float* oo = (l == 11) ? feats : h;             // last layer writes feats output
    gemm(stream, mh16, wT, b2 + o1, h, oo, nullptr, MROWS, DMODEL, DMLP, 0);
  }

  // ---- cross-attention (query = key = value = feats, no layernorm) ----
  { size_t n = (size_t)MROWS * DMODEL;
    k_cast16<<<(int)((n + 255) / 256), 256, 0, stream>>>(feats, ft16, n); }
  cvtT(stream, cwq, wT, DMODEL, DMODEL);
  gemm(stream, ft16, wT, cbq, nullptr, nullptr, Qb, MROWS, DMODEL, DMODEL, F_SQK);
  cvtT(stream, cwk, wT, DMODEL, DMODEL);
  gemm(stream, ft16, wT, cbk, nullptr, nullptr, Kb, MROWS, DMODEL, DMODEL, F_SQK);
  cvtT(stream, cwv, wT, DMODEL, DMODEL);
  gemm(stream, ft16, wT, cbv, nullptr, nullptr, Vtb, MROWS, DMODEL, DMODEL, F_SVT);
  k_attention<<<attnGrid, 256, 0, stream>>>(Qb, Kb, Vtb, ctx16, NTOKS);
  cvtT(stream, cwo, wT, DMODEL, DMODEL);
  gemm(stream, ctx16, wT, cbo, nullptr, nullptr, vf16, MROWS, DMODEL, DMODEL, 0);

  // ---- heads: WMMA for the 768->384 GELU layer, tiny dot kernel for logits ----
  cvtT(stream, fh_w1, wT, DMODEL, 384);
  gemm(stream, ft16, wT, fh_b1, nullptr, nullptr, hid16, MROWS, 384, DMODEL, F_GELU);
  { int n = MROWS * 2; k_head2<<<(n + 255) / 256, 256, 0, stream>>>(hid16, fh_w2, fh_b2, fascia, 2); }
  cvtT(stream, vh_w1, wT, DMODEL, 384);
  gemm(stream, vf16, wT, vh_b1, nullptr, nullptr, hid16, MROWS, 384, DMODEL, F_GELU);
  { int n = MROWS * 4; k_head2<<<(n + 255) / 256, 256, 0, stream>>>(hid16, vh_w2, vh_b2, vein, 4); }
  cvtT(stream, ch_w1, wT, DMODEL, 384);
  gemm(stream, vf16, wT, ch_b1, nullptr, nullptr, hid16, MROWS, 384, DMODEL, F_GELU);
  { int n = MROWS * 3; k_head2<<<(n + 255) / 256, 256, 0, stream>>>(hid16, ch_w2, ch_b2, clsout, 3); }

  (void)in_sizes; (void)n_in; (void)out_size; (void)ws_size;
}
